// CONC_56358560858750
// MI455X (gfx1250) — matrix operand order repaired and checked by reading, not compiled, not run on hardware
//
#include <hip/hip_runtime.h>
#include <math.h>

// ---------- types for WMMA ----------
typedef __attribute__((ext_vector_type(16))) __bf16          v16bf;
typedef __attribute__((ext_vector_type(2)))  __bf16          v2bf;
typedef __attribute__((ext_vector_type(8)))  float           v8f;
typedef __attribute__((ext_vector_type(8)))  unsigned short  v8us;
typedef __attribute__((ext_vector_type(16))) unsigned short  v16us;

__device__ __forceinline__ unsigned short f2bf(float f) {
    // fp32 -> bf16 round-to-nearest-even (manual)
    unsigned u = __float_as_uint(f);
    unsigned r = 0x7FFFu + ((u >> 16) & 1u);
    return (unsigned short)((u + r) >> 16);
}
__device__ __forceinline__ unsigned int f2bf2(float lo, float hi) {
#if __has_builtin(__builtin_amdgcn_cvt_pk_bf16_f32)
    v2bf p = __builtin_amdgcn_cvt_pk_bf16_f32(lo, hi);
    return __builtin_bit_cast(unsigned int, p);
#else
    return (unsigned)f2bf(lo) | ((unsigned)f2bf(hi) << 16);
#endif
}

// ====================================================================
// WMMA GEMM:  xw = A[M,K] @ W[K,Nf]
//   if aggOut != null:  C = xw (raw)  and  aggOut = xw*dis[m]^2 + bias[n]
//   else:               C = xw + bias (+ sigmoid if act==1)
// A fp32 row-major (optional fused ReLU on load), W fp32 row-major;
// converted to bf16 while staging into double-buffered LDS tiles.
// Block: 256 threads = 8 wave32; tile 128(M) x 64(N), K-step 32.
// Wave grid 4(M) x 2(N); each wave computes 32x32 via 2x2 WMMA tiles.
// Single __syncthreads per K-step (ping-pong buffers).
// ====================================================================
#define BM 128
#define BN 64
#define BK 32

__global__ __launch_bounds__(256)
void wmma_gemm_kernel(const float* __restrict__ A, const float* __restrict__ W,
                      const float* __restrict__ bias, float* __restrict__ C,
                      float* __restrict__ aggOut, const float* __restrict__ dis,
                      int M, int K, int Nf, int act, int aRelu)
{
    __shared__ __attribute__((aligned(16))) unsigned short As[2][BM][BK]; // 16 KB [m][k]
    __shared__ __attribute__((aligned(16))) unsigned short Bs[2][BN][BK]; //  8 KB [n][k]^T

    const int tid   = threadIdx.x;
    const int wave  = tid >> 5;
    const int lane  = tid & 31;
    const int waveM = wave & 3;      // 0..3  -> 32-row slab
    const int waveN = wave >> 2;     // 0..1  -> 32-col slab
    const int blockM = blockIdx.x * BM;
    const int blockN = blockIdx.y * BN;

    const int half = lane >> 4;      // 0: lanes 0-15, 1: lanes 16-31
    const int l16  = lane & 15;

    // --- staging coordinates (fixed per thread) ---
    const int ar  = tid >> 1;              // A row 0..127
    const int ac  = (tid & 1) * 16;        // A col 0 or 16
    const int agm = blockM + ar;
    const int bk  = (tid >> 4) * 2;        // B k pair 0,2,...,30
    const int bn0 = (tid & 15) * 4;        // B n group 0,4,...,60

    auto stage = [&](int p, int k0) {
        // A tile 128x32: 2 threads/row, 16 fp32 each -> 2x uint4 LDS stores
        uint4 w0, w1;
        if (agm < M) {
            const float* src = A + (size_t)agm * K + k0 + ac;
            float4 v0 = *(const float4*)(src + 0);
            float4 v1 = *(const float4*)(src + 4);
            float4 v2 = *(const float4*)(src + 8);
            float4 v3 = *(const float4*)(src + 12);
            if (aRelu) {
                v0.x = fmaxf(v0.x, 0.f); v0.y = fmaxf(v0.y, 0.f); v0.z = fmaxf(v0.z, 0.f); v0.w = fmaxf(v0.w, 0.f);
                v1.x = fmaxf(v1.x, 0.f); v1.y = fmaxf(v1.y, 0.f); v1.z = fmaxf(v1.z, 0.f); v1.w = fmaxf(v1.w, 0.f);
                v2.x = fmaxf(v2.x, 0.f); v2.y = fmaxf(v2.y, 0.f); v2.z = fmaxf(v2.z, 0.f); v2.w = fmaxf(v2.w, 0.f);
                v3.x = fmaxf(v3.x, 0.f); v3.y = fmaxf(v3.y, 0.f); v3.z = fmaxf(v3.z, 0.f); v3.w = fmaxf(v3.w, 0.f);
            }
            w0.x = f2bf2(v0.x, v0.y); w0.y = f2bf2(v0.z, v0.w);
            w0.z = f2bf2(v1.x, v1.y); w0.w = f2bf2(v1.z, v1.w);
            w1.x = f2bf2(v2.x, v2.y); w1.y = f2bf2(v2.z, v2.w);
            w1.z = f2bf2(v3.x, v3.y); w1.w = f2bf2(v3.z, v3.w);
        } else {
            w0 = make_uint4(0u, 0u, 0u, 0u);
            w1 = w0;
        }
        *(uint4*)&As[p][ar][ac + 0] = w0;
        *(uint4*)&As[p][ar][ac + 8] = w1;

        // B tile 32x64 transposed: thread = 2 k's x 4 n's, pair-packed u32
        const float* s0 = W + (size_t)(k0 + bk) * Nf + blockN + bn0;
        const float* s1 = s0 + Nf;
#pragma unroll
        for (int i = 0; i < 4; ++i) {
            const int gn = blockN + bn0 + i;
            float u0 = 0.0f, u1 = 0.0f;
            if (gn < Nf) { u0 = s0[i]; u1 = s1[i]; }
            *(unsigned int*)&Bs[p][bn0 + i][bk] = f2bf2(u0, u1);
        }
    };

    v8f acc[2][2] = {};
    const int nk = K / BK;

    stage(0, 0);
    for (int kt = 0; kt < nk; ++kt) {
        __syncthreads();                     // staged tile p visible to all
        if (kt + 1 < nk) stage((kt + 1) & 1, (kt + 1) * BK);  // prefetch other buffer
        const int p = kt & 1;

#pragma unroll
        for (int im = 0; im < 2; ++im) {
            const int m  = waveM * 32 + im * 16 + l16;
            const int ab = half * 8;   // K group {0..7}/{8..15}, +16 for hi half
            v8us alo = *(const v8us*)&As[p][m][ab];
            v8us ahi = *(const v8us*)&As[p][m][16 + ab];
            v16us av = __builtin_shufflevector(alo, ahi,
                        0,1,2,3,4,5,6,7,8,9,10,11,12,13,14,15);
            v16bf a = __builtin_bit_cast(v16bf, av);
#pragma unroll
            for (int in = 0; in < 2; ++in) {
                const int n = waveN * 32 + in * 16 + l16;
                v8us blo = *(const v8us*)&Bs[p][n][half * 8];
                v8us bhi = *(const v8us*)&Bs[p][n][16 + half * 8];
                v16us bv = __builtin_shufflevector(blo, bhi,
                            0,1,2,3,4,5,6,7,8,9,10,11,12,13,14,15);
                v16bf b = __builtin_bit_cast(v16bf, bv);
                acc[im][in] = __builtin_amdgcn_wmma_f32_16x16x32_bf16(
                    false, a, false, b, (short)0, acc[im][in], false, false);
            }
        }
    }

    // ---- fused epilogue ----
    float bv[2];
#pragma unroll
    for (int in = 0; in < 2; ++in) {
        const int gn = blockN + waveN * 32 + in * 16 + l16;
        bv[in] = (bias && gn < Nf) ? bias[gn] : 0.0f;
    }
#pragma unroll
    for (int im = 0; im < 2; ++im) {
#pragma unroll
        for (int r = 0; r < 8; ++r) {
            const int gm = blockM + waveM * 32 + im * 16 + half * 8 + r;
            if (gm >= M) continue;
            float d2 = 0.0f;
            if (aggOut) { const float d = dis[gm]; d2 = d * d; }
#pragma unroll
            for (int in = 0; in < 2; ++in) {
                const int gn = blockN + waveN * 32 + in * 16 + l16;
                if (gn >= Nf) continue;
                const size_t idx = (size_t)gm * Nf + gn;
                float v = acc[im][in][r];
                if (aggOut) {
                    C[idx]      = v;                    // raw xw for edge gather
                    aggOut[idx] = v * d2 + bv[in];      // self-loop + bias init
                } else {
                    v += bv[in];
                    if (act == 1) v = 1.0f / (1.0f + expf(-v));
                    C[idx] = v;
                }
            }
        }
    }
}

// ====================================================================
// Graph kernels
// ====================================================================
__global__ void deg_init_kernel(float* deg, int N) {
    int i = blockIdx.x * blockDim.x + threadIdx.x;
    if (i < N) deg[i] = 1.0f;                 // self-loop
}

__global__ void deg_count_kernel(const long long* __restrict__ col, float* deg, int E) {
    int e = blockIdx.x * blockDim.x + threadIdx.x;
    if (e < E) atomicAdd(&deg[(int)col[e]], 1.0f);
}

__global__ void dis_kernel(float* dis, int N) {
    int i = blockIdx.x * blockDim.x + threadIdx.x;
    if (i < N) dis[i] = rsqrtf(dis[i]);       // in-place deg -> deg^-0.5
}

// one wave32 per edge; lanes cover F with float4; scatter via L2 float atomics
__global__ __launch_bounds__(256)
void edge_scatter_kernel(const float* __restrict__ xw,
                         const long long* __restrict__ row,
                         const long long* __restrict__ col,
                         const float* __restrict__ dis,
                         float* __restrict__ agg, int E, int F)
{
    const int gw   = (int)(((size_t)blockIdx.x * blockDim.x + threadIdx.x) >> 5);
    const int lane = threadIdx.x & 31;
    if (gw >= E) return;
    const int r = (int)row[gw];
    const int c = (int)col[gw];
    const float norm = dis[r] * dis[c];
    const float* src = xw  + (size_t)r * F;
    float*       dst = agg + (size_t)c * F;
    for (int f = lane * 4; f < F; f += 128) {
        float4 v = *(const float4*)(src + f);
        atomicAdd(dst + f + 0, v.x * norm);
        atomicAdd(dst + f + 1, v.y * norm);
        atomicAdd(dst + f + 2, v.z * norm);
        atomicAdd(dst + f + 3, v.w * norm);
    }
}

// ====================================================================
// Host driver
// ====================================================================
extern "C" void kernel_launch(void* const* d_in, const int* in_sizes, int n_in,
                              void* d_out, int out_size, void* d_ws, size_t ws_size,
                              hipStream_t stream)
{
    const float*     x   = (const float*)d_in[0];
    const long long* ei  = (const long long*)d_in[1];
    const int N = in_sizes[0] / 128;
    const int E = in_sizes[1] / 2;
    const long long* row = ei;
    const long long* col = ei + E;

    const float* W1  = (const float*)d_in[2];  const float* b1  = (const float*)d_in[3];
    const float* W2  = (const float*)d_in[4];  const float* b2  = (const float*)d_in[5];
    const float* Wc  = (const float*)d_in[6];  const float* bc  = (const float*)d_in[7];
    const float* We1 = (const float*)d_in[8];  const float* be1 = (const float*)d_in[9];
    const float* We2 = (const float*)d_in[10]; const float* be2 = (const float*)d_in[11];
    const float* We3 = (const float*)d_in[12]; const float* be3 = (const float*)d_in[13];
    const float* We4 = (const float*)d_in[14]; const float* be4 = (const float*)d_in[15];
    const float* Wt  = (const float*)d_in[16]; const float* bt  = (const float*)d_in[17];

    float* ws  = (float*)d_ws;
    float* dis = ws;                               // [N]
    float* B0  = dis + N;                          // [N,256] xw scratch
    float* B1  = B0 + (size_t)N * 256;             // [N,256] agg ping
    float* B2  = B1 + (size_t)N * 256;             // [N,128] agg pong
    float* outLogits = (float*)d_out;              // [N,16]
    float* outTrust  = outLogits + (size_t)N * 16; // [N,2]

    // --- degrees / normalization ---
    deg_init_kernel <<<(N + 255) / 256, 256, 0, stream>>>(dis, N);
    deg_count_kernel<<<(E + 255) / 256, 256, 0, stream>>>(col, dis, E);
    dis_kernel      <<<(N + 255) / 256, 256, 0, stream>>>(dis, N);

    auto gemm = [&](const float* A, const float* W, const float* bias, float* C,
                    float* aggOut, int K, int Nf, int act, int aRelu) {
        dim3 g((N + BM - 1) / BM, (Nf + BN - 1) / BN);
        wmma_gemm_kernel<<<g, 256, 0, stream>>>(A, W, bias, C, aggOut, dis,
                                                N, K, Nf, act, aRelu);
    };
    // GCN layer: fused GEMM (xw -> B0, agg-init -> out) then edge scatter into out
    auto gcn = [&](const float* h, const float* W, const float* b,
                   float* out, int K, int F, int aRelu) {
        gemm(h, W, b, B0, out, K, F, 0, aRelu);
        size_t thr = (size_t)E * 32;
        edge_scatter_kernel<<<(unsigned)((thr + 255) / 256), 256, 0, stream>>>(
            B0, row, col, dis, out, E, F);
    };

    // --- classification branch ---
    gcn(x,  W1, b1, B1, 128, 256, 0);
    gcn(B1, W2, b2, B2, 256, 128, 1);              // ReLU fused into A-staging
    gemm(B2, Wc, bc, outLogits, nullptr, 128, 16, 0, 0);

    // --- encoder branch (4 GCN layers, 128->128) ---
    gcn(x,  We1, be1, B1, 128, 128, 0);
    gcn(B1, We2, be2, B2, 128, 128, 0);
    gcn(B2, We3, be3, B1, 128, 128, 0);
    gcn(B1, We4, be4, B2, 128, 128, 0);

    // --- trust head: sigmoid(h2 @ Wt + bt) ---
    gemm(B2, Wt, bt, outTrust, nullptr, 128, 2, 1, 0);
}
